// OmniAttention_49460843381203
// MI455X (gfx1250) — compile-verified, matrix operand order checked by reading, not run on hardware
//
#include <hip/hip_runtime.h>
#include <hip/hip_bf16.h>

// Problem constants (fixed by the reference)
#define BB 4
#define LL 4096
#define DD 1024
#define HH 16
#define DH 64
#define MTOK (BB * LL)      // 16384 tokens

typedef __attribute__((ext_vector_type(16))) __bf16 v16bf;
typedef __attribute__((ext_vector_type(8)))  __bf16 v8bf;
typedef __attribute__((ext_vector_type(8)))  float  v8f;

// ---- helpers ---------------------------------------------------------------

__device__ __forceinline__ unsigned short f2bf_bits(float f) {
    unsigned int u = __float_as_uint(f);
    // round-to-nearest-even
    unsigned int r = (u + 0x7FFFu + ((u >> 16) & 1u)) >> 16;
    return (unsigned short)r;
}

__device__ __forceinline__ __bf16 f2bf(float f) {
    union { unsigned short u; __bf16 b; } cv;
    cv.u = f2bf_bits(f);
    return cv.b;
}

// Load a 16x32 bf16 A/B fragment (per-lane: two 16-byte contiguous groups,
// lo = K {base..base+7}, hi = K {base+16..base+23}); caller folds in the
// lane-half*8 K offset and the lane-row/col into `p`.
__device__ __forceinline__ v16bf load_frag(const __bf16* p) {
    v8bf lo = *(const v8bf*)(p);
    v8bf hi = *(const v8bf*)(p + 16);
    v16bf r;
#pragma unroll
    for (int i = 0; i < 8; ++i) { r[i] = lo[i]; r[8 + i] = hi[i]; }
    return r;
}

__device__ __forceinline__ v8f wmma_bf16(v16bf a, v16bf b, v8f c) {
    // D = A(16x32) * B(32x16) + C(16x16), fp32 accumulate
    return __builtin_amdgcn_wmma_f32_16x16x32_bf16(
        false, a, false, b, (short)0, c, false, false);
}

// ---- fp32 -> bf16 convert --------------------------------------------------

__global__ __launch_bounds__(256) void cvt_f32_bf16(
    const float* __restrict__ in, __bf16* __restrict__ out, int n) {
    int i = (blockIdx.x * 256 + threadIdx.x) * 4;
    if (i < n) {
        float4 v = *(const float4*)(in + i);
        out[i + 0] = f2bf(v.x);
        out[i + 1] = f2bf(v.y);
        out[i + 2] = f2bf(v.z);
        out[i + 3] = f2bf(v.w);
    }
}

// ---- generic 16x64-per-wave WMMA GEMM:  C[M,N] = A[M,K] * W[N,K]^T ---------
// MODE 0: write bf16 C (row-major, ld=N)
// MODE 1: KV epilogue: N==2048; K part -> Kbuf[token][h*64+d] (bf16),
//         V part -> Vt[b][h][d][l] (bf16, key-contiguous)
// MODE 2: write fp32 C (row-major, ld=N)
template <int MODE>
__global__ __launch_bounds__(256) void gemm16x64(
    const __bf16* __restrict__ A, const __bf16* __restrict__ W,
    __bf16* __restrict__ Cb, float* __restrict__ Cf,
    __bf16* __restrict__ Kout, __bf16* __restrict__ Vt,
    int M, int N, int K) {
    const int lane = threadIdx.x & 31;
    const int widx = threadIdx.x >> 5;
    const int gw   = blockIdx.x * 8 + widx;
    const int nt64 = N >> 6;
    const int mt = gw / nt64;
    const int nt = gw % nt64;
    const int m0 = mt << 4;
    const int n0 = nt << 6;
    if (m0 >= M) return;

    const int lrow = lane & 15;
    const int half = lane >> 4;

    const __bf16* aBase = A + (size_t)(m0 + lrow) * K + half * 8;
    const __bf16* wBase = W + (size_t)(n0 + lrow) * K + half * 8;

    v8f acc[4];
#pragma unroll
    for (int t = 0; t < 4; ++t)
#pragma unroll
        for (int i = 0; i < 8; ++i) acc[t][i] = 0.f;

    for (int kb = 0; kb < K; kb += 32) {
        __builtin_prefetch(aBase + kb + 128, 0, 1);  // global_prefetch_b8
        v16bf a = load_frag(aBase + kb);
#pragma unroll
        for (int t = 0; t < 4; ++t) {
            v16bf b = load_frag(wBase + (size_t)t * 16 * K + kb);
            acc[t] = wmma_bf16(a, b, acc[t]);
        }
    }

#pragma unroll
    for (int t = 0; t < 4; ++t) {
        const int col = n0 + 16 * t + lrow;
#pragma unroll
        for (int r = 0; r < 8; ++r) {
            const int row = m0 + r + half * 8;
            const float v = acc[t][r];
            if (MODE == 0) {
                Cb[(size_t)row * N + col] = f2bf(v);
            } else if (MODE == 2) {
                Cf[(size_t)row * N + col] = v;
            } else {
                const int h = col >> 7;        // 128 cols per head (K|V)
                const int rem = col & 127;
                if (rem < DH) {                // K component
                    Kout[(size_t)row * DD + h * DH + rem] = f2bf(v);
                } else {                       // V component -> transposed
                    const int d = rem - DH;
                    const int b = row >> 12;   // token / LL
                    const int l = row & (LL - 1);
                    Vt[((size_t)((b * HH + h) * DH + d)) * LL + l] = f2bf(v);
                }
            }
        }
    }
}

// ---- flash attention: one 16-query tile per wave ---------------------------
// Q: [token][h*64+d] bf16 ; Kbuf: [token][h*64+d] bf16 ; Vt: [b][h][d][l] bf16
// O: [token][h*64+d] bf16
//
// Softmax engineering:
//  * running row-sums are produced by a 5th WMMA against a ones-column B
//    fragment (l = P @ 1), rescaled online exactly like the O tiles -> no
//    cross-lane sum reductions in the hot loop.
//  * the softmax shift uses a shared max over the 8 rows owned by this
//    lane-half (a valid per-row upper bound -> exact softmax), so the
//    max reduction is 15 VALU fmax + one 4-step butterfly per chunk.
__global__ __launch_bounds__(256) void flash_attn(
    const __bf16* __restrict__ Q, const __bf16* __restrict__ Kb,
    const __bf16* __restrict__ Vt, __bf16* __restrict__ O) {
    __shared__ __align__(16) unsigned short Pshm[8][16 * 32];  // per-wave P tile

    const int lane = threadIdx.x & 31;
    const int widx = threadIdx.x >> 5;
    const int gt   = blockIdx.x * 8 + widx;       // global 16-query tile id
    const int bh   = gt >> 8;                     // LL/16 = 256 tiles per (b,h)
    const int qt   = gt & 255;
    const int b    = bh >> 4;
    const int h    = bh & 15;
    const int m0   = qt << 4;
    const int lrow = lane & 15;
    const int half = lane >> 4;

    // Q fragments for this tile (K-dim = DH = 64 -> two 16x32 fragments)
    const size_t qoff = (size_t)(b * LL + m0 + lrow) * DD + h * DH + half * 8;
    const v16bf qa0 = load_frag(Q + qoff);
    const v16bf qa1 = load_frag(Q + qoff + 32);

    // ones-column B fragment: B[k][0] = 1 for all k, else 0
    v16bf onesb;
    {
        const __bf16 one  = f2bf(1.0f);
        const __bf16 zero = f2bf(0.0f);
#pragma unroll
        for (int i = 0; i < 16; ++i) onesb[i] = (lrow == 0) ? one : zero;
    }

    // o[0..3] = output tiles (d chunks of 16); o[4] col 0 = running row-sums
    v8f o[5];
#pragma unroll
    for (int t = 0; t < 5; ++t)
#pragma unroll
        for (int i = 0; i < 8; ++i) o[t][i] = 0.f;

    float mrun = -1e30f;   // running shift (shared by this half's 8 rows)

    unsigned short* P = &Pshm[widx][0];

    for (int j0 = 0; j0 < LL; j0 += 32) {
        // ---- S = Q * K^T for 32 keys (two 16x16 C tiles) ----
        v8f s[2];
#pragma unroll
        for (int tn = 0; tn < 2; ++tn) {
            const __bf16* kbase =
                Kb + (size_t)(b * LL + j0 + 16 * tn + lrow) * DD + h * DH + half * 8;
            v16bf b0 = load_frag(kbase);        // d = 0..31
            v16bf b1 = load_frag(kbase + 32);   // d = 32..63
            v8f c;
#pragma unroll
            for (int i = 0; i < 8; ++i) c[i] = 0.f;
            c = wmma_bf16(qa0, b0, c);
            c = wmma_bf16(qa1, b1, c);
            s[tn] = c;
        }

        // ---- scale + shared-max (half-tile) ----
        float sv0[8], sv1[8];
        float tmax = -1e30f;
#pragma unroll
        for (int r = 0; r < 8; ++r) {
            sv0[r] = s[0][r] * 0.125f;          // 1/sqrt(64)
            sv1[r] = s[1][r] * 0.125f;
            tmax = fmaxf(tmax, fmaxf(sv0[r], sv1[r]));
        }
#pragma unroll
        for (int d = 1; d < 16; d <<= 1)
            tmax = fmaxf(tmax, __shfl_xor(tmax, d, 32));

        const float mnew   = fmaxf(mrun, tmax);
        const float scalef = __expf(mrun - mnew);
        mrun = mnew;

        // ---- rescale accumulators (incl. row-sum tile) ----
#pragma unroll
        for (int t = 0; t < 5; ++t)
#pragma unroll
            for (int r = 0; r < 8; ++r) o[t][r] *= scalef;

        // ---- P = exp(S - m), stash into LDS in bf16 (C layout, 16x32) ----
#pragma unroll
        for (int r = 0; r < 8; ++r) {
            const float p0 = __expf(sv0[r] - mnew);
            const float p1 = __expf(sv1[r] - mnew);
            const int row = r + half * 8;
            P[row * 32 + lrow]      = f2bf_bits(p0);
            P[row * 32 + 16 + lrow] = f2bf_bits(p1);
        }
        asm volatile("s_wait_dscnt 0" ::: "memory");  // LDS in-order per wave

        // ---- reload P in A-fragment layout ----
        const v16bf pa =
            load_frag(reinterpret_cast<const __bf16*>(P + lrow * 32 + half * 8));

        // ---- O += P * V (V transposed -> key-contiguous B fragments) ----
#pragma unroll
        for (int t = 0; t < 4; ++t) {
            const __bf16* vb =
                Vt + ((size_t)((b * HH + h) * DH + 16 * t + lrow)) * LL + j0 + half * 8;
            v16bf bv = load_frag(vb);
            o[t] = wmma_bf16(pa, bv, o[t]);
        }
        // ---- l += P * ones (row-sums ride the matrix pipe) ----
        o[4] = wmma_bf16(pa, onesb, o[4]);
    }

    // ---- normalize and write out ----
#pragma unroll
    for (int r = 0; r < 8; ++r) {
        // row-sum for this row lives in column 0 of tile 4 (lane 0 / lane 16)
        const float ls  = __shfl(o[4][r], lane & 16, 32);
        const float inv = 1.f / ls;
        const int row = m0 + r + half * 8;
#pragma unroll
        for (int t = 0; t < 4; ++t) {
            O[(size_t)(b * LL + row) * DD + h * DH + 16 * t + lrow] =
                f2bf(o[t][r] * inv);
        }
    }
}

// ---- host glue -------------------------------------------------------------

extern "C" void kernel_launch(void* const* d_in, const int* in_sizes, int n_in,
                              void* d_out, int out_size, void* d_ws, size_t ws_size,
                              hipStream_t stream) {
    const float* x    = (const float*)d_in[0];   // (B,L,D)
    const float* Wq   = (const float*)d_in[1];   // (D,D)
    const float* Wkv  = (const float*)d_in[2];   // (2D,D)
    const float* Wout = (const float*)d_in[3];   // (D,D)
    float* out = (float*)d_out;

    const size_t e_x    = (size_t)MTOK * DD;     // 16,777,216
    const size_t e_wq   = (size_t)DD * DD;
    const size_t e_wkv  = (size_t)2 * DD * DD;
    const size_t e_wout = (size_t)DD * DD;

    __bf16* xb    = (__bf16*)d_ws;
    __bf16* wqb   = xb    + e_x;
    __bf16* wkvb  = wqb   + e_wq;
    __bf16* woutb = wkvb  + e_wkv;
    __bf16* qb    = woutb + e_wout;
    __bf16* kbuf  = qb    + e_x;
    __bf16* vtb   = kbuf  + e_x;
    __bf16* ab    = vtb   + e_x;

    // fp32 -> bf16 (4 elems/thread, 256 threads/block)
    cvt_f32_bf16<<<(int)(e_x    / 1024), 256, 0, stream>>>(x,    xb,    (int)e_x);
    cvt_f32_bf16<<<(int)(e_wq   / 1024), 256, 0, stream>>>(Wq,   wqb,   (int)e_wq);
    cvt_f32_bf16<<<(int)(e_wkv  / 1024), 256, 0, stream>>>(Wkv,  wkvb,  (int)e_wkv);
    cvt_f32_bf16<<<(int)(e_wout / 1024), 256, 0, stream>>>(Wout, woutb, (int)e_wout);

    // Q = x * Wq^T  (16384x1024x1024) -> bf16
    gemm16x64<0><<<(MTOK / 16) * (DD / 64) / 8, 256, 0, stream>>>(
        xb, wqb, qb, nullptr, nullptr, nullptr, MTOK, DD, DD);

    // KV = x * Wkv^T (16384x2048x1024) -> K token-major bf16, V transposed bf16
    gemm16x64<1><<<(MTOK / 16) * (2 * DD / 64) / 8, 256, 0, stream>>>(
        xb, wkvb, nullptr, nullptr, kbuf, vtb, MTOK, 2 * DD, DD);

    // fused attention -> ab (bf16)
    flash_attn<<<(BB * HH * (LL / 16)) / 8, 256, 0, stream>>>(qb, kbuf, vtb, ab);

    // out = attn * Wout^T -> fp32 d_out
    gemm16x64<2><<<(MTOK / 16) * (DD / 64) / 8, 256, 0, stream>>>(
        ab, woutb, nullptr, out, nullptr, nullptr, MTOK, DD, DD);
}
